// GLM4Attention_89172111000195
// MI455X (gfx1250) — compile-verified, matrix operand order checked by reading, not run on hardware
//
#include <hip/hip_runtime.h>

// ---------------------------------------------------------------- constants
constexpr int B_ = 2, S_ = 2048, HID_ = 4096;
constexpr int H_ = 32, KV_ = 8, D_ = 128;
constexpr int G_ = H_ / KV_;
constexpr float EPS_ = 1e-5f;

typedef __attribute__((ext_vector_type(4)))  float        f32x4;
typedef __attribute__((ext_vector_type(8)))  float        f32x8;
typedef __attribute__((ext_vector_type(2)))  unsigned int u32x2;
typedef __attribute__((ext_vector_type(4)))  unsigned int u32x4;
typedef __attribute__((ext_vector_type(8)))  unsigned int u32x8;
typedef __attribute__((ext_vector_type(16))) __bf16       bf16x16;

// ---------------------------------------------------------------- helpers
__device__ inline unsigned short f2bf(float f) {
  unsigned int u = __float_as_uint(f);
  u += 0x7FFFu + ((u >> 16) & 1u);           // round-to-nearest-even
  return (unsigned short)(u >> 16);
}

__device__ inline f32x8 zero8() {
  f32x8 v;
#pragma unroll
  for (int i = 0; i < 8; ++i) v[i] = 0.0f;
  return v;
}

// One 16x32 bf16 WMMA operand fragment (ISA 16-bit A layout):
// lane holds row = lane%16; K chunks at half*8 and 16+half*8 within the
// 32-wide K block. `p` = row start at the desired K block; `half` = lane/16.
__device__ inline bf16x16 frag16x32(const unsigned short* p, int half) {
  union { u32x4 q[2]; bf16x16 v; } u;
  u.q[0] = *(const u32x4*)(p + half * 8);
  u.q[1] = *(const u32x4*)(p + 16 + half * 8);
  return u.v;
}

__device__ inline f32x8 wmma_bf16(bf16x16 a, bf16x16 b, f32x8 c) {
  return __builtin_amdgcn_wmma_f32_16x16x32_bf16(false, a, false, b, (short)0, c,
                                                 false, false);
}

// stage 4 elements into bf16 LDS
__device__ inline void stage4(unsigned short* dst, const float* src) {
  f32x4 v = *(const f32x4*)src;
  u32x2 o;
  o[0] = (unsigned int)f2bf(v[0]) | ((unsigned int)f2bf(v[1]) << 16);
  o[1] = (unsigned int)f2bf(v[2]) | ((unsigned int)f2bf(v[3]) << 16);
  *(u32x2*)dst = o;
}
__device__ inline void stage4(unsigned short* dst, const unsigned short* src) {
  *(u32x2*)dst = *(const u32x2*)src;
}

// ---------------------------------------------------------------- GEMM
// C[M,N] (f32) = A[M,K] * Bw[N,K]^T   (A,Bw converted to bf16 on the fly)
template <typename TA, typename TB>
__global__ __launch_bounds__(256) void gemm_wmma_k(const TA* __restrict__ A,
                                                   const TB* __restrict__ Bw,
                                                   float* __restrict__ C,
                                                   int M, int N, int K) {
  constexpr int LDT = 40;  // padded bf16 row stride (32 + 8)
  __shared__ unsigned short As[128 * LDT];
  __shared__ unsigned short Bs[128 * LDT];

  const int tid = threadIdx.x;
  const int wave = tid >> 5, lane = tid & 31;
  const int lrow = lane & 15, half = lane >> 4;
  const int wm = wave >> 2, wn = wave & 3;  // 2x4 wave grid -> 64x32 per wave
  const int m0 = blockIdx.y * 128, n0 = blockIdx.x * 128;

  f32x8 acc[4][2];
#pragma unroll
  for (int mi = 0; mi < 4; ++mi)
#pragma unroll
    for (int ni = 0; ni < 2; ++ni) acc[mi][ni] = zero8();

  const int ldrow = tid >> 3;      // 0..31
  const int ldc4 = (tid & 7) * 4;  // 0,4,...,28

  for (int kt = 0; kt < K; kt += 32) {
#pragma unroll
    for (int p = 0; p < 4; ++p) {
      const int row = ldrow + p * 32;
      stage4(&As[row * LDT + ldc4], A + (size_t)(m0 + row) * K + kt + ldc4);
      stage4(&Bs[row * LDT + ldc4], Bw + (size_t)(n0 + row) * K + kt + ldc4);
    }
    if (kt + 32 < K) {
      __builtin_prefetch(A + (size_t)(m0 + ldrow) * K + kt + 32 + ldc4, 0, 3);
      __builtin_prefetch(Bw + (size_t)(n0 + ldrow) * K + kt + 32 + ldc4, 0, 3);
    }
    __syncthreads();

    bf16x16 af[4], bfr[2];
#pragma unroll
    for (int mi = 0; mi < 4; ++mi)
      af[mi] = frag16x32(&As[(wm * 64 + mi * 16 + lrow) * LDT], half);
#pragma unroll
    for (int ni = 0; ni < 2; ++ni)
      bfr[ni] = frag16x32(&Bs[(wn * 32 + ni * 16 + lrow) * LDT], half);
#pragma unroll
    for (int mi = 0; mi < 4; ++mi)
#pragma unroll
      for (int ni = 0; ni < 2; ++ni)
        acc[mi][ni] = wmma_bf16(af[mi], bfr[ni], acc[mi][ni]);
    __syncthreads();
  }

#pragma unroll
  for (int mi = 0; mi < 4; ++mi)
#pragma unroll
    for (int ni = 0; ni < 2; ++ni)
#pragma unroll
      for (int r = 0; r < 8; ++r) {
        const int row = m0 + wm * 64 + mi * 16 + r + half * 8;
        const int col = n0 + wn * 32 + ni * 16 + lrow;
        C[(size_t)row * N + col] = acc[mi][ni][r];
      }
}

// ------------------------------------------------- RMS-norm + partial RoPE
// src f32 [b][s][nH][d]  ->  dst bf16 [b][nH][s][d]; one wave per row.
__global__ __launch_bounds__(128) void norm_rope_k(const float* __restrict__ src,
                                                   const float* __restrict__ w,
                                                   unsigned short* __restrict__ dst,
                                                   int nH) {
  const int wave = threadIdx.x >> 5, lane = threadIdx.x & 31;
  const long row = (long)blockIdx.x * 4 + wave;  // (b*nH + h)*S + s
  const int s = (int)(row % S_);
  const int h = (int)((row / S_) % nH);
  const int b = (int)(row / ((long)S_ * nH));
  const float* sp = src + (((size_t)b * S_ + s) * nH + h) * D_;

  float x0 = sp[lane];
  float x1 = sp[lane + 32];
  float x2 = sp[lane + 64];
  float x3 = sp[lane + 96];

  float ssq = x0 * x0 + x1 * x1 + x2 * x2 + x3 * x3;
#pragma unroll
  for (int off = 1; off < 32; off <<= 1) ssq += __shfl_xor(ssq, off, 32);
  const float inv = rsqrtf(ssq * (1.0f / (float)D_) + EPS_);

  x0 *= inv * w[lane];
  x1 *= inv * w[lane + 32];
  x2 *= inv * w[lane + 64];
  x3 *= inv * w[lane + 96];

  // RoPE over first RD=64 dims; pair (lane, lane+32) is in-lane.
  // inv_freq = 1e6^(-lane/32) = exp(-lane/32 * ln(1e6))
  const float ang = (float)s * __expf(-(float)lane * (13.815510557964274f / 32.0f));
  const float c = __cosf(ang), sn = __sinf(ang);
  const float y0 = x0 * c - x1 * sn;
  const float y1 = x1 * c + x0 * sn;

  unsigned short* dp = dst + (size_t)row * D_;
  dp[lane]      = f2bf(y0);
  dp[lane + 32] = f2bf(y1);
  dp[lane + 64] = f2bf(x2);
  dp[lane + 96] = f2bf(x3);
}

// ------------------------------------------- V: f32 [b][s][kv][d] -> bf16 [b][kv][s][d]
__global__ __launch_bounds__(256) void vconv_k(const float* __restrict__ src,
                                               unsigned short* __restrict__ dst) {
  const size_t i = (size_t)blockIdx.x * 256 + threadIdx.x;  // dst-flat index
  const int d = (int)(i & (D_ - 1));
  const int s = (int)((i >> 7) & (S_ - 1));
  const size_t bk = i >> 18;  // S_*D_ == 2^18
  const int kv = (int)(bk & (KV_ - 1));
  const int b = (int)(bk >> 3);
  dst[i] = f2bf(src[(((size_t)b * S_ + s) * KV_ + kv) * D_ + d]);
}

// ---------------------------------------------------------------- attention
constexpr int KLD = 136;  // K tile row stride (64 keys x 128 d) = 128 + 8 pad
constexpr int VLD = 72;   // V^T tile row stride (128 d x 64 keys)
constexpr int PLD = 72;   // per-wave probs strip (16 x 64)

__global__ __launch_bounds__(128) void attn_k(const unsigned short* __restrict__ qbf,
                                              const unsigned short* __restrict__ kbf,
                                              const unsigned short* __restrict__ vbf,
                                              const float* __restrict__ mask,
                                              unsigned short* __restrict__ aout) {
  __shared__ unsigned short Kt[64 * KLD];
  __shared__ unsigned short Vt[128 * VLD];
  __shared__ unsigned short Pt[4 * 16 * PLD];

  const int tid = threadIdx.x;
  const int wave = tid >> 5, lane = tid & 31;
  const int lrow = lane & 15, half = lane >> 4;

  const int qb = blockIdx.x, h = blockIdx.y, b = blockIdx.z;
  const int kv = h / G_;
  const int q0 = qb * 64;

  const unsigned short* Qb = qbf + (((size_t)(b * H_ + h)) * S_ + q0 + wave * 16) * D_;
  const unsigned short* Kb = kbf + ((size_t)(b * KV_ + kv)) * S_ * D_;
  const unsigned short* Vb = vbf + ((size_t)(b * KV_ + kv)) * S_ * D_;
  const float* Mb = mask + (size_t)b * S_ * S_ + (size_t)q0 * S_;

  // Q fragments for this wave's 16 rows: D=128 -> 4 K-chunks of 32
  bf16x16 qf[4];
#pragma unroll
  for (int c = 0; c < 4; ++c) qf[c] = frag16x32(Qb + lrow * D_ + c * 32, half);

  f32x8 oacc[8];
#pragma unroll
  for (int db = 0; db < 8; ++db) oacc[db] = zero8();
  float rowmax[8], rowsum[8];
#pragma unroll
  for (int r = 0; r < 8; ++r) { rowmax[r] = -3.0e38f; rowsum[r] = 0.0f; }

  unsigned short* P = &Pt[wave * 16 * PLD];
  const unsigned kt_lds = (unsigned)(size_t)(void*)&Kt[0];  // LDS byte address

  for (int kb = 0; kb < S_ / 64; ++kb) {
    const int k0 = kb * 64;

    // --- K tile via Tensor Data Mover: 64x128 bf16 tile of the [2048,128]
    //     K-cache DMA'd straight into LDS, with TDM padding producing the
    //     272-byte (128+8 elem) row stride the fragment loads expect.
    if (wave == 0) {
      const unsigned long long ga =
          (unsigned long long)(size_t)(const void*)(Kb + (size_t)k0 * D_);
      u32x4 g0;
      g0[0] = 1u;                                   // count=1 (valid D#)
      g0[1] = kt_lds;                               // lds_addr
      g0[2] = (unsigned)ga;                         // global_addr[31:0]
      g0[3] = (unsigned)((ga >> 32) & 0x01FFFFFFu)  // global_addr[56:32]
              | (2u << 30);                         // type=2 ("image")
      u32x8 g1;
      g1[0] = (1u << 16)        // data_size = 2 bytes
            | (1u << 20)        // pad_enable
            | (5u << 22)        // pad_interval = 64 DWORDs (one 128-elem row)
            | (3u << 25);       // pad_amount   = 4 DWORDs (8 elems)
      g1[1] = ((unsigned)D_ & 0xFFFFu) << 16;  // tensor_dim0[15:0] = 128
      g1[2] = ((unsigned)S_ & 0xFFFFu) << 16;  // tensor_dim1[15:0] = 2048
      g1[3] = ((unsigned)D_) << 16;            // tile_dim0 = 128
      g1[4] = 64u;                             // tile_dim1 = 64, tile_dim2 = 0
      g1[5] = (unsigned)D_;                    // tensor_dim0_stride = 128
      g1[6] = 0u;
      g1[7] = 0u;
      u32x4 g2 = {0u, 0u, 0u, 0u};
      u32x4 g3 = {0u, 0u, 0u, 0u};
      asm volatile("tensor_load_to_lds %0, %1, %2, %3"
                   :
                   : "s"(g0), "s"(g1), "s"(g2), "s"(g3)
                   : "memory");
    }

    // --- V tile transposed into Vt by all waves (overlaps the tensor DMA)
#pragma unroll
    for (int p = 0; p < 8; ++p) {
      const int idx = p * 128 + tid;  // 1024 chunks of 8 bf16
      const int row = idx >> 4;       // key 0..63
      const int c8 = idx & 15;        // d-chunk 0..15
      u32x4 dv = *(const u32x4*)(Vb + (size_t)(k0 + row) * D_ + c8 * 8);
      unsigned short t[8];
      *(u32x4*)t = dv;
#pragma unroll
      for (int e = 0; e < 8; ++e) Vt[(c8 * 8 + e) * VLD + row] = t[e];
    }
    if (wave == 0) __builtin_amdgcn_s_wait_tensorcnt(0);
    __syncthreads();

    // scores: 16 q-rows x 64 keys
    f32x8 sacc[4];
#pragma unroll
    for (int nb = 0; nb < 4; ++nb) sacc[nb] = zero8();
#pragma unroll
    for (int nb = 0; nb < 4; ++nb)
#pragma unroll
      for (int c = 0; c < 4; ++c) {
        bf16x16 kf = frag16x32(&Kt[(nb * 16 + lrow) * KLD + c * 32], half);
        sacc[nb] = wmma_bf16(qf[c], kf, sacc[nb]);
      }

    const float sc = 0.088388347648318447f;  // 1/sqrt(128)
#pragma unroll
    for (int nb = 0; nb < 4; ++nb)
#pragma unroll
      for (int r = 0; r < 8; ++r) {
        const float mv =
            Mb[(size_t)(wave * 16 + r + half * 8) * S_ + k0 + nb * 16 + lrow];
        sacc[nb][r] = sacc[nb][r] * sc + mv;
      }

    // online softmax; row r lives in a 16-lane half-wave
#pragma unroll
    for (int r = 0; r < 8; ++r) {
      float mx = fmaxf(fmaxf(sacc[0][r], sacc[1][r]), fmaxf(sacc[2][r], sacc[3][r]));
#pragma unroll
      for (int off = 1; off < 16; off <<= 1) mx = fmaxf(mx, __shfl_xor(mx, off, 32));
      const float nm = fmaxf(rowmax[r], mx);
      const float fac = __expf(rowmax[r] - nm);
      rowmax[r] = nm;
      float ps = 0.0f;
#pragma unroll
      for (int nb = 0; nb < 4; ++nb) {
        const float pv = __expf(sacc[nb][r] - nm);
        sacc[nb][r] = pv;
        ps += pv;
      }
#pragma unroll
      for (int off = 1; off < 16; off <<= 1) ps += __shfl_xor(ps, off, 32);
      rowsum[r] = rowsum[r] * fac + ps;
#pragma unroll
      for (int db = 0; db < 8; ++db) oacc[db][r] *= fac;
    }

    // probs -> wave-private LDS strip -> A fragments (same-wave LDS is in-order)
#pragma unroll
    for (int nb = 0; nb < 4; ++nb)
#pragma unroll
      for (int r = 0; r < 8; ++r)
        P[(r + half * 8) * PLD + nb * 16 + lrow] = f2bf(sacc[nb][r]);

    bf16x16 pf[2];
#pragma unroll
    for (int c = 0; c < 2; ++c) pf[c] = frag16x32(&P[lrow * PLD + c * 32], half);
#pragma unroll
    for (int db = 0; db < 8; ++db)
#pragma unroll
      for (int c = 0; c < 2; ++c) {
        bf16x16 vf = frag16x32(&Vt[(db * 16 + lrow) * VLD + c * 32], half);
        oacc[db] = wmma_bf16(pf[c], vf, oacc[db]);
      }
    __syncthreads();
  }

  // normalize + write bf16 to [b][s][h][d] (= [B*S, H*D] for the O-proj)
#pragma unroll
  for (int r = 0; r < 8; ++r) {
    const float rinv = 1.0f / rowsum[r];
    const int s = q0 + wave * 16 + r + half * 8;
#pragma unroll
    for (int db = 0; db < 8; ++db)
      aout[(((size_t)b * S_ + s) * H_ + h) * D_ + db * 16 + lrow] =
          f2bf(oacc[db][r] * rinv);
  }
}

// ---------------------------------------------------------------- launch
extern "C" void kernel_launch(void* const* d_in, const int* in_sizes, int n_in,
                              void* d_out, int out_size, void* d_ws, size_t ws_size,
                              hipStream_t stream) {
  (void)in_sizes; (void)n_in; (void)out_size; (void)ws_size;
  const float* hs   = (const float*)d_in[0];
  const float* mask = (const float*)d_in[1];
  const float* wq   = (const float*)d_in[2];
  const float* wk   = (const float*)d_in[3];
  const float* wv   = (const float*)d_in[4];
  const float* wo   = (const float*)d_in[5];
  const float* qnw  = (const float*)d_in[6];
  const float* knw  = (const float*)d_in[7];
  float* out = (float*)d_out;
  char* ws = (char*)d_ws;

  const size_t qfB = (size_t)B_ * S_ * H_ * D_ * sizeof(float);   // 64 MiB
  const size_t kfB = (size_t)B_ * S_ * KV_ * D_ * sizeof(float);  // 16 MiB
  const size_t qbB = (size_t)B_ * H_ * S_ * D_ * 2;               // 32 MiB
  const size_t kbB = (size_t)B_ * KV_ * S_ * D_ * 2;              //  8 MiB

  float* qf = (float*)(ws);
  float* kf = (float*)(ws + qfB);
  float* vf = (float*)(ws + qfB + kfB);
  unsigned short* qb = (unsigned short*)(ws + qfB + 2 * kfB);
  unsigned short* kb = (unsigned short*)(ws + qfB + 2 * kfB + qbB);
  unsigned short* vb = (unsigned short*)(ws + qfB + 2 * kfB + qbB + kbB);
  unsigned short* ao = (unsigned short*)(ws);  // aliases qf (dead by then)

  // Q/K/V projections (bf16 WMMA, f32 accumulate)
  gemm_wmma_k<float, float><<<dim3((H_ * D_) / 128, (B_ * S_) / 128), 256, 0,
                              stream>>>(hs, wq, qf, B_ * S_, H_ * D_, HID_);
  gemm_wmma_k<float, float><<<dim3((KV_ * D_) / 128, (B_ * S_) / 128), 256, 0,
                              stream>>>(hs, wk, kf, B_ * S_, KV_ * D_, HID_);
  gemm_wmma_k<float, float><<<dim3((KV_ * D_) / 128, (B_ * S_) / 128), 256, 0,
                              stream>>>(hs, wv, vf, B_ * S_, KV_ * D_, HID_);

  // per-head RMS-norm + partial RoPE, transpose to [b,h,s,d] bf16
  norm_rope_k<<<(B_ * H_ * S_) / 4, 128, 0, stream>>>(qf, qnw, qb, H_);
  norm_rope_k<<<(B_ * KV_ * S_) / 4, 128, 0, stream>>>(kf, knw, kb, KV_);
  vconv_k<<<((size_t)B_ * S_ * KV_ * D_) / 256, 256, 0, stream>>>(vf, vb);

  // flash-style attention (TDM K-tile + WMMA)
  attn_k<<<dim3(S_ / 64, H_, B_), 128, 0, stream>>>(qb, kb, vb, mask, ao);

  // output projection
  gemm_wmma_k<unsigned short, float><<<dim3(HID_ / 128, (B_ * S_) / 128), 256, 0,
                                       stream>>>(ao, wo, out, B_ * S_, HID_, H_ * D_);
}
